// SparseFusion_22299470201620
// MI455X (gfx1250) — matrix operand order repaired and checked by reading, not compiled, not run on hardware
//
#include <hip/hip_runtime.h>
#include <hip/hip_bf16.h>

#define N_VOXC  100000
#define K3C     125
#define IMG_CC  64
#define PTS_CC  32
#define OUT_CC  16
#define EPSC    1e-5f

typedef __bf16 bf16_t;
typedef __attribute__((ext_vector_type(16))) __bf16 v16bf;
typedef __attribute__((ext_vector_type(8)))  __bf16 v8bf;
typedef __attribute__((ext_vector_type(8)))  float  v8f;

// ---------------------------------------------------------------------------
// One wave32 accumulates a 32x16 (voxel x out-channel) tile as two 16x16
// WMMA C-fragments that SHARE each B fragment (halves weight traffic, gives
// back-to-back v_wmma issue). v_wmma_f32_16x16x32_bf16, f32 accumulation.
//
// A fragment (16x32 bf16, ISA 7.12.2): lane L holds row M = L&15.
//   lanes 0-15 : K 0..15 ; lanes 16-31: K 16..31 (contiguous, same order as B)
//   -> two contiguous 16B (v8bf) loads per lane per K-step.
// B fragment (32x16 bf16): lane L holds column N = L&15;
//   lanes 0-15 K=0..15, lanes 16-31 K=16..31 (VGPR j -> K 2j,2j+1)
//   -> weights pre-transposed to [k][cout][cin] bf16: one contiguous 32B load.
// Gather offsets are 32-bit (idx*CIN*2 < 13MB) so the backend can use
// saddr + 32-bit voffset global addressing instead of 64-bit VALU adds.
// ---------------------------------------------------------------------------
template<int CIN>
__device__ __forceinline__ void conv_tile_acc2(const bf16_t* __restrict__ featsb,
                                               const bf16_t* __restrict__ wbt,
                                               const int*    __restrict__ nbr,
                                               int n0, int lane,
                                               v8f& acc0, v8f& acc1) {
    const int col  = lane & 15;        // row for A, column for B/C
    const int hi   = lane >> 4;        // which half of the wave
    const int asub = hi * 16;          // A K-subblock select
    const int kb   = hi * 16;          // B K-half select
    const int* nrow0 = nbr + (size_t)(n0 + col) * K3C;
    const int* nrow1 = nbr + (size_t)(n0 + 16 + col) * K3C;
    const char* fbase = (const char*)featsb;

    for (int k = 0; k < K3C; ++k) {
        const unsigned off0 = (unsigned)nrow0[k] * (unsigned)(CIN * 2);
        const unsigned off1 = (unsigned)nrow1[k] * (unsigned)(CIN * 2);
        const bf16_t* wb = wbt + ((size_t)k * OUT_CC + col) * CIN + kb;
#pragma unroll
        for (int c0 = 0; c0 < CIN; c0 += 32) {
            const v16bf b = *(const v16bf*)(wb + c0);
            const unsigned cb = (unsigned)((c0 + asub) * 2);

            v8bf a0lo = *(const v8bf*)(fbase + off0 + cb);
            v8bf a0hi = *(const v8bf*)(fbase + off0 + cb + 16);
            v16bf a0;
#pragma unroll
            for (int i = 0; i < 8; ++i) { a0[i] = a0lo[i]; a0[8 + i] = a0hi[i]; }
            acc0 = __builtin_amdgcn_wmma_f32_16x16x32_bf16(
                false, a0, false, b, (short)0, acc0, false, false);

            v8bf a1lo = *(const v8bf*)(fbase + off1 + cb);
            v8bf a1hi = *(const v8bf*)(fbase + off1 + cb + 16);
            v16bf a1;
#pragma unroll
            for (int i = 0; i < 8; ++i) { a1[i] = a1lo[i]; a1[8 + i] = a1hi[i]; }
            acc1 = __builtin_amdgcn_wmma_f32_16x16x32_bf16(
                false, a1, false, b, (short)0, acc1, false, false);
        }
    }
}

// conv + BN + ReLU, writes f32 branch output and its bf16 slice of cat[:,32]
template<int CIN, int CATOFF>
__global__ __launch_bounds__(256) void sparse_conv_bnrelu(
    const bf16_t* __restrict__ featsb, const bf16_t* __restrict__ wbt,
    const int*    __restrict__ nbr,
    const float*  __restrict__ bn_g, const float* __restrict__ bn_b,
    const float*  __restrict__ bn_m, const float* __restrict__ bn_v,
    float* __restrict__ outf, bf16_t* __restrict__ catb)
{
    const int lane = threadIdx.x & 31;
    const int wave = threadIdx.x >> 5;
    const int tile = blockIdx.x * 8 + wave;
    const int n_tiles = (N_VOXC + 31) / 32;
    if (tile >= n_tiles) return;                 // wave-uniform: EXEC stays all-1s
    const int n0 = tile * 32;

    v8f acc0 = {}, acc1 = {};
    conv_tile_acc2<CIN>(featsb, wbt, nbr, n0, lane, acc0, acc1);

    const int col = lane & 15;
    const int hi  = lane >> 4;
    const float sc = bn_g[col] * rsqrtf(bn_v[col] + EPSC);
    const float bi = bn_b[col] - bn_m[col] * sc;
#pragma unroll
    for (int r = 0; r < 8; ++r) {                // C VGPR r -> row r (+8 upper half)
        const int rw0 = n0 + hi * 8 + r;
        const int rw1 = rw0 + 16;
        const float y0 = fmaxf(acc0[r] * sc + bi, 0.0f);
        const float y1 = fmaxf(acc1[r] * sc + bi, 0.0f);
        outf[(size_t)rw0 * OUT_CC + col] = y0;
        outf[(size_t)rw1 * OUT_CC + col] = y1;
        catb[(size_t)rw0 * 32 + CATOFF + col] = (bf16_t)y0;
        catb[(size_t)rw1 * 32 + CATOFF + col] = (bf16_t)y1;
    }
}

// vis conv + BN + ReLU + row-wise 16-dot + sigmoid + final blend
__global__ __launch_bounds__(256) void sparse_conv_vis(
    const bf16_t* __restrict__ catb, const bf16_t* __restrict__ wbt,
    const int*    __restrict__ nbr,
    const float*  __restrict__ bn_g, const float* __restrict__ bn_b,
    const float*  __restrict__ bn_m, const float* __restrict__ bn_v,
    const float*  __restrict__ wv2,
    const float*  __restrict__ imgf, const float* __restrict__ ptsf,
    float* __restrict__ out)
{
    const int lane = threadIdx.x & 31;
    const int wave = threadIdx.x >> 5;
    const int tile = blockIdx.x * 8 + wave;
    const int n_tiles = (N_VOXC + 31) / 32;
    if (tile >= n_tiles) return;
    const int n0 = tile * 32;

    v8f acc0 = {}, acc1 = {};
    conv_tile_acc2<32>(catb, wbt, nbr, n0, lane, acc0, acc1);

    const int col = lane & 15;
    const int hi  = lane >> 4;
    const float sc = bn_g[col] * rsqrtf(bn_v[col] + EPSC);
    const float bi = bn_b[col] - bn_m[col] * sc;
    const float wc = wv2[col];
#pragma unroll
    for (int r = 0; r < 8; ++r) {
        // subtile 0
        {
            const float y = fmaxf(acc0[r] * sc + bi, 0.0f);
            float s = y * wc;                     // reduce across the 16 lanes of this half
            s += __shfl_xor(s, 8, 32);
            s += __shfl_xor(s, 4, 32);
            s += __shfl_xor(s, 2, 32);
            s += __shfl_xor(s, 1, 32);
            const float vw = 1.0f / (1.0f + __expf(-s));
            const size_t o = (size_t)(n0 + hi * 8 + r) * OUT_CC + col;
            out[o] = vw * imgf[o] + (1.0f - vw) * ptsf[o];
        }
        // subtile 1
        {
            const float y = fmaxf(acc1[r] * sc + bi, 0.0f);
            float s = y * wc;
            s += __shfl_xor(s, 8, 32);
            s += __shfl_xor(s, 4, 32);
            s += __shfl_xor(s, 2, 32);
            s += __shfl_xor(s, 1, 32);
            const float vw = 1.0f / (1.0f + __expf(-s));
            const size_t o = (size_t)(n0 + 16 + hi * 8 + r) * OUT_CC + col;
            out[o] = vw * imgf[o] + (1.0f - vw) * ptsf[o];
        }
    }
}

// ---------------------------------------------------------------------------
// Prep kernels
// ---------------------------------------------------------------------------
__global__ void cvt_feats_bf16(const float* __restrict__ src, bf16_t* __restrict__ dst,
                               int n_valid, int total) {
    int i = blockIdx.x * 256 + threadIdx.x;
    if (i >= total) return;
    dst[i] = (i < n_valid) ? (bf16_t)src[i] : (bf16_t)0.0f;   // sentinel row -> 0
}

// src [K3][cin][16] f32  ->  dst [K3][16][cin] bf16
__global__ void cvt_weights_t(const float* __restrict__ src, bf16_t* __restrict__ dst, int cin) {
    int i = blockIdx.x * 256 + threadIdx.x;
    int total = K3C * cin * OUT_CC;
    if (i >= total) return;
    int ci = i % cin;
    int t  = i / cin;
    int co = t % OUT_CC;
    int k  = t / OUT_CC;
    dst[i] = (bf16_t)src[((size_t)k * cin + ci) * OUT_CC + co];
}

__global__ void zero_cat_sentinel(bf16_t* __restrict__ catb) {
    if (threadIdx.x < 32) catb[(size_t)N_VOXC * 32 + threadIdx.x] = (bf16_t)0.0f;
}

// ---------------------------------------------------------------------------
extern "C" void kernel_launch(void* const* d_in, const int* in_sizes, int n_in,
                              void* d_out, int out_size, void* d_ws, size_t ws_size,
                              hipStream_t stream) {
    const float* img_f   = (const float*)d_in[0];
    const float* pts_f   = (const float*)d_in[1];
    const int*   nbr     = (const int*)  d_in[2];
    const float* W_img   = (const float*)d_in[3];
    const float* big     = (const float*)d_in[4];
    const float* bib     = (const float*)d_in[5];
    const float* bim     = (const float*)d_in[6];
    const float* biv     = (const float*)d_in[7];
    const float* W_pts   = (const float*)d_in[8];
    const float* bpg     = (const float*)d_in[9];
    const float* bpb     = (const float*)d_in[10];
    const float* bpm     = (const float*)d_in[11];
    const float* bpv     = (const float*)d_in[12];
    const float* W_vis1  = (const float*)d_in[13];
    const float* bvg     = (const float*)d_in[14];
    const float* bvb     = (const float*)d_in[15];
    const float* bvm     = (const float*)d_in[16];
    const float* bvv     = (const float*)d_in[17];
    const float* W_vis2  = (const float*)d_in[18];
    float* out = (float*)d_out;

    // workspace carve-up (256B aligned)
    char* ws = (char*)d_ws;
    size_t off = 0;
    auto take = [&](size_t bytes) { char* p = ws + off; off = (off + bytes + 255) & ~(size_t)255; return p; };
    bf16_t* imgb  = (bf16_t*)take((size_t)(N_VOXC + 1) * IMG_CC * sizeof(bf16_t));
    bf16_t* ptsb  = (bf16_t*)take((size_t)(N_VOXC + 1) * PTS_CC * sizeof(bf16_t));
    bf16_t* catb  = (bf16_t*)take((size_t)(N_VOXC + 1) * 32     * sizeof(bf16_t));
    float*  imgo  = (float*) take((size_t)N_VOXC * OUT_CC * sizeof(float));
    float*  ptso  = (float*) take((size_t)N_VOXC * OUT_CC * sizeof(float));
    bf16_t* wimgt = (bf16_t*)take((size_t)K3C * OUT_CC * IMG_CC * sizeof(bf16_t));
    bf16_t* wptst = (bf16_t*)take((size_t)K3C * OUT_CC * PTS_CC * sizeof(bf16_t));
    bf16_t* wvist = (bf16_t*)take((size_t)K3C * OUT_CC * PTS_CC * sizeof(bf16_t));
    (void)ws_size; (void)in_sizes; (void)n_in; (void)out_size;

    // --- prep: weight transpose+convert, feature convert, cat sentinel ---
    {
        int t = K3C * IMG_CC * OUT_CC;
        cvt_weights_t<<<(t + 255) / 256, 256, 0, stream>>>(W_img, wimgt, IMG_CC);
        t = K3C * PTS_CC * OUT_CC;
        cvt_weights_t<<<(t + 255) / 256, 256, 0, stream>>>(W_pts,  wptst, PTS_CC);
        cvt_weights_t<<<(t + 255) / 256, 256, 0, stream>>>(W_vis1, wvist, PTS_CC);

        int tot = (N_VOXC + 1) * IMG_CC;
        cvt_feats_bf16<<<(tot + 255) / 256, 256, 0, stream>>>(img_f, imgb, N_VOXC * IMG_CC, tot);
        tot = (N_VOXC + 1) * PTS_CC;
        cvt_feats_bf16<<<(tot + 255) / 256, 256, 0, stream>>>(pts_f, ptsb, N_VOXC * PTS_CC, tot);
        zero_cat_sentinel<<<1, 64, 0, stream>>>(catb);
    }

    // --- WMMA convs: 1 wave = 32x16 tile (two 16x16 C fragments) ---
    const int n_tiles = (N_VOXC + 31) / 32;           // 3125
    dim3 grid((n_tiles + 7) / 8), block(256);

    sparse_conv_bnrelu<IMG_CC, 0><<<grid, block, 0, stream>>>(
        imgb, wimgt, nbr, big, bib, bim, biv, imgo, catb);
    sparse_conv_bnrelu<PTS_CC, 16><<<grid, block, 0, stream>>>(
        ptsb, wptst, nbr, bpg, bpb, bpm, bpv, ptso, catb);

    sparse_conv_vis<<<grid, block, 0, stream>>>(
        catb, wvist, nbr, bvg, bvb, bvm, bvv, W_vis2, imgo, ptso, out);
}
